// GCN_27977416966469
// MI455X (gfx1250) — compile-verified, matrix operand order checked by reading, not run on hardware
//
#include <hip/hip_runtime.h>
#include <hip/hip_bf16.h>

typedef __attribute__((ext_vector_type(16))) _Float16 v16h;
typedef __attribute__((ext_vector_type(8)))  float    v8f;

#define GCN_D_IN  512
#define GCN_D_HID 256
#define GCN_D_OUT 64

// ---------------------------------------------------------------- degree prep
__global__ void gcn_init_deg(float* __restrict__ deg, int n) {
    int i = blockIdx.x * blockDim.x + threadIdx.x;
    if (i < n) deg[i] = 1.0f;  // self-loop contributes 1 to every node
}

__global__ void gcn_accum_deg(const int* __restrict__ dst, float* __restrict__ deg, int E) {
    int t = blockIdx.x * blockDim.x + threadIdx.x;
    if (t < E) atomicAdd(&deg[dst[t]], 1.0f);
}

__global__ void gcn_make_dinv(float* __restrict__ deg, int n) {
    int i = blockIdx.x * blockDim.x + threadIdx.x;
    if (i < n) {
        float d = deg[i];
        deg[i] = (d > 0.0f) ? rsqrtf(d) : 0.0f;  // in-place deg -> dinv
    }
}

// ---------------------------------------------------------------- weight pack
// Pack W[K,N] (fp32 row-major) into f16 WMMA B-fragments:
//   fragment f = kt*(N/16)+tn is 32 lanes x 16 halves (1KB), lane-contiguous.
//   lane L: col n = tn*16 + (L&15); k = kt*32 + (L>>4)*8 + j + (j>=8 ? 8 : 0)
// One thread per (fragment, lane): 16 strided reads -> 32B contiguous write.
__global__ void gcn_pack_w_f16(const float* __restrict__ W, _Float16* __restrict__ Wp,
                               int K, int N) {
    int t = blockIdx.x * blockDim.x + threadIdx.x;
    int tilesN = N >> 4;
    int total = (K >> 5) * tilesN * 32;
    if (t >= total) return;
    int frag = t >> 5;
    int lane = t & 31;
    int kt = frag / tilesN;
    int tn = frag - kt * tilesN;
    int n    = tn * 16 + (lane & 15);
    int koff = kt * 32 + (lane >> 4) * 8;
    _Float16* o = Wp + (size_t)t * 16;
#pragma unroll
    for (int j = 0; j < 8; ++j) {
        o[j]     = (_Float16)W[(size_t)(koff + j) * N + n];
        o[j + 8] = (_Float16)W[(size_t)(koff + 16 + j) * N + n];
    }
}

// ---------------------------------------------------------------- WMMA GEMM
// C[M,N] = A[M,K] (fp32 row-major) x B (pre-packed f16 fragments).
// One wave32 computes a 16x64 slab: A fragment loaded+converted once per
// k-step, reused across 4 v_wmma_f32_16x16x32_f16. Requires K%32==0, N%64==0.
__global__ void gcn_wmma_gemm_f16x4(const float* __restrict__ A,
                                    const _Float16* __restrict__ Bp,
                                    float* __restrict__ C,
                                    int M, int K, int N) {
    const int lane    = threadIdx.x & 31;
    const int wave    = (blockIdx.x * blockDim.x + threadIdx.x) >> 5;
    const int tilesN  = N >> 4;   // 16-wide tiles
    const int groupsN = N >> 6;   // 64-wide groups of 4 tiles
    const int tm = wave / groupsN;
    const int tg = wave - tm * groupsN;
    if (tm * 16 >= M) return;
    const int tn0  = tg * 4;
    const int hsel = lane >> 4;   // 0: K {0..7,16..23}; 1: {8..15,24..31}
    const int idx  = lane & 15;   // A row within tile

    const float* __restrict__ arow = A + (size_t)(tm * 16 + idx) * K;

    v8f acc0 = {}, acc1 = {}, acc2 = {}, acc3 = {};
    for (int k0 = 0; k0 < K; k0 += 32) {
        const int kb = k0 + hsel * 8;

        // A fragment: k = kb..kb+7 and kb+16..kb+23 (ISA 16-bit A layout)
        float4 a01 = *(const float4*)(arow + kb);
        float4 a23 = *(const float4*)(arow + kb + 4);
        float4 a45 = *(const float4*)(arow + kb + 16);
        float4 a67 = *(const float4*)(arow + kb + 20);
        if (k0 + 32 < K) __builtin_prefetch(arow + kb + 32, 0, 3);

        float af[16];
        *(float4*)(af + 0)  = a01;  *(float4*)(af + 4)  = a23;
        *(float4*)(af + 8)  = a45;  *(float4*)(af + 12) = a67;
        v16h av;
#pragma unroll
        for (int j = 0; j < 16; ++j) av[j] = (_Float16)af[j];

        // 4 pre-packed B fragments: 32B contiguous per lane each
        const int kt = k0 >> 5;
        const _Float16* bbase = Bp + ((size_t)(kt * tilesN + tn0) * 32 + lane) * 16;
        v16h b0 = *(const v16h*)(bbase);
        v16h b1 = *(const v16h*)(bbase + 1 * 32 * 16);
        v16h b2 = *(const v16h*)(bbase + 2 * 32 * 16);
        v16h b3 = *(const v16h*)(bbase + 3 * 32 * 16);

        acc0 = __builtin_amdgcn_wmma_f32_16x16x32_f16(false, av, false, b0, (short)0, acc0, false, false);
        acc1 = __builtin_amdgcn_wmma_f32_16x16x32_f16(false, av, false, b1, (short)0, acc1, false, false);
        acc2 = __builtin_amdgcn_wmma_f32_16x16x32_f16(false, av, false, b2, (short)0, acc2, false, false);
        acc3 = __builtin_amdgcn_wmma_f32_16x16x32_f16(false, av, false, b3, (short)0, acc3, false, false);
    }

    // C/D layout: VGPR r holds row r (lanes 0-15) / row r+8 (lanes 16-31)
    float* cbase = C + (size_t)(tm * 16 + hsel * 8) * N + tn0 * 16 + idx;
#pragma unroll
    for (int r = 0; r < 8; ++r) {
        cbase[(size_t)r * N + 0]  = acc0[r];
        cbase[(size_t)r * N + 16] = acc1[r];
        cbase[(size_t)r * N + 32] = acc2[r];
        cbase[(size_t)r * N + 48] = acc3[r];
    }
}

// ---------------------------------------------------------------- aggregation
// out[i,f] = xw[i,f] * dinv[i]^2   (self-loop term; fully initializes out)
__global__ void gcn_self_loop(const float* __restrict__ xw,
                              const float* __restrict__ dinv,
                              float* __restrict__ out, int total, int F) {
    int t = blockIdx.x * blockDim.x + threadIdx.x;
    if (t < total) {
        int i = t / F;
        float di = dinv[i];
        out[t] = xw[t] * di * di;
    }
}

// out[d] += xw[s] * dinv[s]*dinv[d]; one thread per (edge, 4-feature chunk)
__global__ void gcn_edge_scatter(const int* __restrict__ src,
                                 const int* __restrict__ dst,
                                 const float* __restrict__ dinv,
                                 const float* __restrict__ xw,
                                 float* __restrict__ out,
                                 int E, int F) {
    int t = blockIdx.x * blockDim.x + threadIdx.x;
    int chunks = F >> 2;
    int e = t / chunks;
    if (e >= E) return;
    int c = t - e * chunks;
    int s = src[e], d = dst[e];
    float norm = dinv[s] * dinv[d];
    float4 v = ((const float4*)(xw + (size_t)s * F))[c];
    float* o = out + (size_t)d * F + 4 * c;
    atomicAdd(o + 0, v.x * norm);
    atomicAdd(o + 1, v.y * norm);
    atomicAdd(o + 2, v.z * norm);
    atomicAdd(o + 3, v.w * norm);
}

__global__ void gcn_bias_relu(float* __restrict__ h, const float* __restrict__ bias,
                              int total, int F) {
    int t = blockIdx.x * blockDim.x + threadIdx.x;
    if (t < total) h[t] = fmaxf(h[t] + bias[t % F], 0.0f);
}

// ---------------------------------------------------------------- softmax
// One wave32 per node over 64 features (2 per lane); wave32 butterfly reduce.
__global__ void gcn_softmax64(const float* __restrict__ h,
                              const float* __restrict__ bias,
                              float* __restrict__ out, int nodes) {
    int wave = (blockIdx.x * blockDim.x + threadIdx.x) >> 5;
    int lane = threadIdx.x & 31;
    if (wave >= nodes) return;
    const float* row = h + (size_t)wave * 64;
    float v0 = row[lane]      + bias[lane];
    float v1 = row[lane + 32] + bias[lane + 32];
    float m = fmaxf(v0, v1);
#pragma unroll
    for (int off = 16; off > 0; off >>= 1) m = fmaxf(m, __shfl_xor(m, off, 32));
    float e0 = __expf(v0 - m);
    float e1 = __expf(v1 - m);
    float s = e0 + e1;
#pragma unroll
    for (int off = 16; off > 0; off >>= 1) s += __shfl_xor(s, off, 32);
    float inv = 1.0f / s;
    float* orow = out + (size_t)wave * 64;
    orow[lane]      = e0 * inv;
    orow[lane + 32] = e1 * inv;
}

// ---------------------------------------------------------------- launcher
extern "C" void kernel_launch(void* const* d_in, const int* in_sizes, int n_in,
                              void* d_out, int out_size, void* d_ws, size_t ws_size,
                              hipStream_t stream) {
    const float* x   = (const float*)d_in[0];
    const int*   ei  = (const int*)  d_in[1];
    const float* W1  = (const float*)d_in[2];
    const float* b1  = (const float*)d_in[3];
    const float* W2  = (const float*)d_in[4];
    const float* b2  = (const float*)d_in[5];
    float* out = (float*)d_out;

    const int N = in_sizes[0] / GCN_D_IN;   // 50000
    const int E = in_sizes[1] / 2;          // 800000
    const int* src = ei;
    const int* dst = ei + E;

    // Workspace (floats), 256B-aligned slices, peak ~103 MB:
    //   dinv [N] | region [N*D_HID] (xw1, later xw2+h2) | h1 [N*D_HID]
    //   | W1p (f16, 512*256) | W2p (f16, 256*64)
    float* ws   = (float*)d_ws;
    size_t off  = ((size_t)N + 63) & ~(size_t)63;
    float* dinv = ws;
    float* xw1  = ws + off;
    float* h1   = xw1 + (size_t)N * GCN_D_HID;
    _Float16* W1p = (_Float16*)(h1 + (size_t)N * GCN_D_HID);
    _Float16* W2p = W1p + (size_t)GCN_D_IN * GCN_D_HID;
    float* xw2  = xw1;                            // reuse: xw1 dead after layer-1 scatter
    float* h2   = xw1 + (size_t)N * GCN_D_OUT;    // disjoint from xw2 slice

    const int B = 256;
    // --- pack weights to f16 B-fragments (tiny; independent of everything else)
    {
        int t1 = (GCN_D_IN >> 5) * (GCN_D_HID >> 4) * 32;
        gcn_pack_w_f16<<<(t1 + B - 1) / B, B, 0, stream>>>(W1, W1p, GCN_D_IN, GCN_D_HID);
        int t2 = (GCN_D_HID >> 5) * (GCN_D_OUT >> 4) * 32;
        gcn_pack_w_f16<<<(t2 + B - 1) / B, B, 0, stream>>>(W2, W2p, GCN_D_HID, GCN_D_OUT);
    }

    // --- degrees -> dinv
    gcn_init_deg <<<(N + B - 1) / B, B, 0, stream>>>(dinv, N);
    gcn_accum_deg<<<(E + B - 1) / B, B, 0, stream>>>(dst, dinv, E);
    gcn_make_dinv<<<(N + B - 1) / B, B, 0, stream>>>(dinv, N);

    // --- layer 1: xw1 = x @ W1 (WMMA, 16x64 slab per wave), aggregate, bias+relu
    {
        int waves = (N / 16) * (GCN_D_HID / 64);
        int th = waves * 32;
        gcn_wmma_gemm_f16x4<<<(th + B - 1) / B, B, 0, stream>>>(x, W1p, xw1, N, GCN_D_IN, GCN_D_HID);
    }
    {
        int total = N * GCN_D_HID;
        gcn_self_loop<<<(total + B - 1) / B, B, 0, stream>>>(xw1, dinv, h1, total, GCN_D_HID);
        int et = E * (GCN_D_HID / 4);
        gcn_edge_scatter<<<(et + B - 1) / B, B, 0, stream>>>(src, dst, dinv, xw1, h1, E, GCN_D_HID);
        gcn_bias_relu<<<(total + B - 1) / B, B, 0, stream>>>(h1, b1, total, GCN_D_HID);
    }

    // --- layer 2: xw2 = h1 @ W2 (WMMA), aggregate, bias+softmax
    {
        int waves = (N / 16) * (GCN_D_OUT / 64);
        int th = waves * 32;
        gcn_wmma_gemm_f16x4<<<(th + B - 1) / B, B, 0, stream>>>(h1, W2p, xw2, N, GCN_D_HID, GCN_D_OUT);
    }
    {
        int total = N * GCN_D_OUT;
        gcn_self_loop<<<(total + B - 1) / B, B, 0, stream>>>(xw2, dinv, h2, total, GCN_D_OUT);
        int et = E * (GCN_D_OUT / 4);
        gcn_edge_scatter<<<(et + B - 1) / B, B, 0, stream>>>(src, dst, dinv, xw2, h2, E, GCN_D_OUT);
    }
    {
        int th = N * 32;  // one wave per node
        gcn_softmax64<<<(th + B - 1) / B, B, 0, stream>>>(h2, b2, out, N);
    }
}
